// Graphormer_31327491457422
// MI455X (gfx1250) — matrix operand order repaired and checked by reading, not compiled, not run on hardware
//
#include <hip/hip_runtime.h>
#include <hip/hip_bf16.h>
#include <math.h>

// ---------------------------------------------------------------------------
// Types for CDNA5 WMMA (wave32): bf16 A/B fragments, f32 accumulator.
// ---------------------------------------------------------------------------
typedef __bf16 bf16_t;
typedef __attribute__((ext_vector_type(16))) __bf16 v16bf;
typedef __attribute__((ext_vector_type(8)))  float  v8f;

struct alignas(16) U128 { unsigned int x, y, z, w; };
union Frag16 { v16bf v; U128 q[2]; };
struct alignas(8) B16x4 { bf16_t v[4]; };

#define HID 96
#define NHEAD 8
#define HDIM 12
#define FFN_DIM 384
#define INV_SCALE 0.28867513459481287f  // 1/sqrt(12)

// ---------------------------------------------------------------------------
// Zero fp32 buffer (grid-stride, float4). n4 = count of float4 groups.
// ---------------------------------------------------------------------------
__global__ __launch_bounds__(256) void zero_f32(float4* __restrict__ p, int n4) {
    int i = blockIdx.x * blockDim.x + threadIdx.x;
    int stride = gridDim.x * blockDim.x;
    float4 z = make_float4(0.f, 0.f, 0.f, 0.f);
    for (; i < n4; i += stride) p[i] = z;
}

// ---------------------------------------------------------------------------
// Pack a row-major fp32 weight W[k*Nsrc + n] (K x Nsrc) into the WMMA
// B-fragment layout: Wpk[((nt*KS + ks)*32 + lane)*16 + h], where
//   n = nt*16 + (lane & 15)
//   k = ks*32 + ((lane >= 16) ? 8 : 0) + (h < 8 ? h : h + 8)  (ISA 7.12.2, 16-bit B)
// ---------------------------------------------------------------------------
__global__ __launch_bounds__(256) void pack_w(const float* __restrict__ W,
                                              bf16_t* __restrict__ Wpk,
                                              int K, int Nsrc) {
    int idx = blockIdx.x * blockDim.x + threadIdx.x;
    int total = K * Nsrc;
    if (idx >= total) return;
    int KS   = K >> 5;
    int h    = idx & 15;
    int lane = (idx >> 4) & 31;
    int ks   = (idx >> 9) % KS;
    int nt   = idx / (KS << 9);
    int n = nt * 16 + (lane & 15);
    int k = ks * 32 + ((lane >= 16) ? 8 : 0) + (h < 8 ? h : h + 8);
    Wpk[idx] = (bf16_t)W[k * Nsrc + n];
}

// Concatenate bq,bk,bv into a 288-entry bias vector.
__global__ void concat_bias3(const float* a, const float* b, const float* c,
                             float* __restrict__ o) {
    int i = threadIdx.x;
    if (i < 96)       o[i] = a[i];
    else if (i < 192) o[i] = b[i - 96];
    else if (i < 288) o[i] = c[i - 192];
}

// ---------------------------------------------------------------------------
// LayerNorm over rows of 96 fp32 values -> bf16 output. One wave32 per row,
// 3 elements per lane, butterfly reductions via __shfl_xor (warpSize == 32).
// ---------------------------------------------------------------------------
__global__ __launch_bounds__(256) void ln_kernel(const float* __restrict__ x,
                                                 const float* __restrict__ g,
                                                 const float* __restrict__ b,
                                                 bf16_t* __restrict__ y, int nrows) {
    int lane = threadIdx.x & 31;
    int row  = blockIdx.x * 8 + (threadIdx.x >> 5);
    if (row >= nrows) return;
    int base = row * HID;
    float v0 = x[base + lane], v1 = x[base + lane + 32], v2 = x[base + lane + 64];
    float s = v0 + v1 + v2;
#pragma unroll
    for (int o = 16; o > 0; o >>= 1) s += __shfl_xor(s, o, 32);
    float mu = s * (1.0f / 96.0f);
    float d0 = v0 - mu, d1 = v1 - mu, d2 = v2 - mu;
    float sq = d0 * d0 + d1 * d1 + d2 * d2;
#pragma unroll
    for (int o = 16; o > 0; o >>= 1) sq += __shfl_xor(sq, o, 32);
    float rstd = rsqrtf(sq * (1.0f / 96.0f) + 1e-5f);
    y[base + lane]      = (bf16_t)(d0 * rstd * g[lane]      + b[lane]);
    y[base + lane + 32] = (bf16_t)(d1 * rstd * g[lane + 32] + b[lane + 32]);
    y[base + lane + 64] = (bf16_t)(d2 * rstd * g[lane + 64] + b[lane + 64]);
}

// ---------------------------------------------------------------------------
// WMMA GEMM: out[M x Nout] = A[M x K] * Wpk + bias (+epilogue).
// K = KSTEPS*32. Block = 256 threads = 8 waves; each wave owns a 16-row
// stripe (128 rows per block) and sweeps all 16-col tiles, reusing its A
// fragments. A-fragment per lane (ISA 7.12.2, 16-bit A 16x32):
//   row = mbase + (lane&15); halves 0-7  = K {0..7  | 8..15},
//                            halves 8-15 = K {16..23|24..31}  (by lane half)
// EPI: 0 = +bias -> f32 ; 1 = +bias + residual -> f32 ; 2 = +bias, GELU -> bf16
// All indices are 32-bit so stores lower to SADDR + voffset forms.
// ---------------------------------------------------------------------------
template <int KSTEPS, int EPI>
__global__ __launch_bounds__(256) void gemm_wmma(const bf16_t* __restrict__ A,
                                                 const bf16_t* __restrict__ Wpk,
                                                 const float* __restrict__ bias,
                                                 const float* __restrict__ resid,
                                                 float* __restrict__ outf,
                                                 bf16_t* __restrict__ outb,
                                                 int M, int Nout) {
    const int K   = KSTEPS * 32;
    int lane  = threadIdx.x & 31;
    int wave  = threadIdx.x >> 5;
    int mbase = blockIdx.x * 128 + wave * 16;
    int rowA  = mbase + (lane & 15);
    if (rowA >= M) rowA = M - 1;  // clamp loads; stores are guarded
    const bool full = (mbase + 16 <= M);   // wave-uniform

    const bf16_t* arow = A + rowA * K;
    int khalf = (lane < 16) ? 0 : 8;

    Frag16 afrag[KSTEPS];
#pragma unroll
    for (int ks = 0; ks < KSTEPS; ++ks) {
        int k0 = ks * 32 + khalf;
        afrag[ks].q[0] = *(const U128*)(arow + k0);
        afrag[ks].q[1] = *(const U128*)(arow + k0 + 16);
    }

    const int rlo = (lane < 16) ? 0 : 8;   // first of 8 rows this lane owns
    int NT = Nout >> 4;
    for (int nt = 0; nt < NT; ++nt) {
        v8f c = {};
#pragma unroll
        for (int ks = 0; ks < KSTEPS; ++ks) {
            Frag16 bfrag;
            const bf16_t* bp = Wpk + (((nt * KSTEPS + ks) << 5) + lane) * 16;
            bfrag.q[0] = *(const U128*)(bp);
            bfrag.q[1] = *(const U128*)(bp + 8);
            c = __builtin_amdgcn_wmma_f32_16x16x32_bf16(
                    false, afrag[ks].v, false, bfrag.v, (short)0, c, false, false);
        }
        int col  = nt * 16 + (lane & 15);
        float bb = bias[col];
        int base = (mbase + rlo) * Nout + col;

        if (full) {
#pragma unroll
            for (int r = 0; r < 8; ++r) {
                int idx = base + r * Nout;
                float v = c[r] + bb;
                if (EPI == 0) {
                    outf[idx] = v;
                } else if (EPI == 1) {
                    outf[idx] = v + resid[idx];
                } else {  // exact GELU (approximate=False) -> bf16
                    float gv = 0.5f * v * (1.0f + erff(v * 0.70710678118654752f));
                    outb[idx] = (bf16_t)gv;
                }
            }
        } else {
#pragma unroll
            for (int r = 0; r < 8; ++r) {
                int row = mbase + rlo + r;
                if (row < M) {
                    int idx = row * Nout + col;
                    float v = c[r] + bb;
                    if (EPI == 0) {
                        outf[idx] = v;
                    } else if (EPI == 1) {
                        outf[idx] = v + resid[idx];
                    } else {
                        float gv = 0.5f * v * (1.0f + erff(v * 0.70710678118654752f));
                        outb[idx] = (bf16_t)gv;
                    }
                }
            }
        }
    }
}

// ---------------------------------------------------------------------------
// Sparse attention: one thread per (edge, head). qkv[N x 288] holds q|k|v.
// Rows are 16B aligned (48*h bytes), so gather with float4 (b128 loads).
// score = clip(<k[src],q[dst]>/sqrt(D), +-5); exp; scatter-add v[src]*score
// and score into wV / Z with fp32 global atomics (segment_sum semantics).
// ---------------------------------------------------------------------------
__global__ __launch_bounds__(256) void edge_kernel(const int* __restrict__ ei,
                                                   const float* __restrict__ qkv,
                                                   float* __restrict__ wV,
                                                   float* __restrict__ Z, int E) {
    int t = blockIdx.x * blockDim.x + threadIdx.x;
    if (t >= E * NHEAD) return;
    int e = t >> 3, h = t & 7;
    int src = ei[e];
    int dst = ei[E + e];
    const float4* q4 = (const float4*)(qkv + dst * 288 + h * HDIM);
    const float4* k4 = (const float4*)(qkv + src * 288 + 96 + h * HDIM);
    const float4* v4 = (const float4*)(qkv + src * 288 + 192 + h * HDIM);
    float4 qa = q4[0], qb = q4[1], qc = q4[2];
    float4 ka = k4[0], kb = k4[1], kc = k4[2];
    float4 va = v4[0], vb = v4[1], vc = v4[2];
    float s = qa.x * ka.x + qa.y * ka.y + qa.z * ka.z + qa.w * ka.w
            + qb.x * kb.x + qb.y * kb.y + qb.z * kb.z + qb.w * kb.w
            + qc.x * kc.x + qc.y * kc.y + qc.z * kc.z + qc.w * kc.w;
    s *= INV_SCALE;
    s = fminf(fmaxf(s, -5.0f), 5.0f);
    float ex = __expf(s);
    float* wrow = wV + dst * HID + h * HDIM;
    atomicAdd(wrow + 0,  va.x * ex);
    atomicAdd(wrow + 1,  va.y * ex);
    atomicAdd(wrow + 2,  va.z * ex);
    atomicAdd(wrow + 3,  va.w * ex);
    atomicAdd(wrow + 4,  vb.x * ex);
    atomicAdd(wrow + 5,  vb.y * ex);
    atomicAdd(wrow + 6,  vb.z * ex);
    atomicAdd(wrow + 7,  vb.w * ex);
    atomicAdd(wrow + 8,  vc.x * ex);
    atomicAdd(wrow + 9,  vc.y * ex);
    atomicAdd(wrow + 10, vc.z * ex);
    atomicAdd(wrow + 11, vc.w * ex);
    atomicAdd(&Z[dst * NHEAD + h], ex);
}

// wV / (Z + 1e-6) -> bf16 for the O-projection GEMM. One thread per 4 cols;
// a 4-col group never crosses a head boundary (4 divides 12).
__global__ __launch_bounds__(256) void attn_fin(const float* __restrict__ wV,
                                                const float* __restrict__ Z,
                                                bf16_t* __restrict__ abf, int nrows) {
    int g = blockIdx.x * blockDim.x + threadIdx.x;
    if (g >= nrows * (HID / 4)) return;
    int row = g / (HID / 4);
    int c4  = (g % (HID / 4)) * 4;
    float4 w = *(const float4*)(wV + row * HID + c4);
    float inv = 1.0f / (Z[row * NHEAD + c4 / HDIM] + 1e-6f);
    B16x4 o;
    o.v[0] = (bf16_t)(w.x * inv);
    o.v[1] = (bf16_t)(w.y * inv);
    o.v[2] = (bf16_t)(w.z * inv);
    o.v[3] = (bf16_t)(w.w * inv);
    *(B16x4*)(abf + row * HID + c4) = o;
}

// ---------------------------------------------------------------------------
// Launcher
// ---------------------------------------------------------------------------
extern "C" void kernel_launch(void* const* d_in, const int* in_sizes, int n_in,
                              void* d_out, int out_size, void* d_ws, size_t ws_size,
                              hipStream_t stream) {
    const float* x     = (const float*)d_in[0];
    const int*   ei    = (const int*)d_in[1];
    const float* ln1_g = (const float*)d_in[2];
    const float* ln1_b = (const float*)d_in[3];
    const float* Wq    = (const float*)d_in[4];
    const float* bq    = (const float*)d_in[5];
    const float* Wk    = (const float*)d_in[6];
    const float* bk    = (const float*)d_in[7];
    const float* Wv    = (const float*)d_in[8];
    const float* bv    = (const float*)d_in[9];
    const float* Wo    = (const float*)d_in[10];
    const float* bo    = (const float*)d_in[11];
    const float* ln2_g = (const float*)d_in[12];
    const float* ln2_b = (const float*)d_in[13];
    const float* W1    = (const float*)d_in[14];
    const float* b1    = (const float*)d_in[15];
    const float* W2    = (const float*)d_in[16];
    const float* b2    = (const float*)d_in[17];

    const int N = in_sizes[0] / HID;       // 50000
    const int E = in_sizes[1] / 2;         // 800000
    float* out = (float*)d_out;

    // ---- workspace carve-out (256B aligned) ----
    char* ws = (char*)d_ws;
    size_t off = 0;
    auto alloc = [&](size_t bytes) -> char* {
        off = (off + 255) & ~(size_t)255;
        char* p = ws + off;
        off += bytes;
        return p;
    };
    bf16_t* ybf   = (bf16_t*)alloc((size_t)N * HID * 2);
    float*  qkv   = (float*) alloc((size_t)N * 288 * 4);
    float*  wV    = (float*) alloc((size_t)N * HID * 4);   // kept adjacent to Z
    float*  Zacc  = (float*) alloc((size_t)N * NHEAD * 4);
    bf16_t* abf   = (bf16_t*)alloc((size_t)N * HID * 2);
    float*  x2    = (float*) alloc((size_t)N * HID * 4);
    bf16_t* y2bf  = (bf16_t*)alloc((size_t)N * HID * 2);
    bf16_t* g1bf  = (bf16_t*)alloc((size_t)N * FFN_DIM * 2);
    bf16_t* pkQKV = (bf16_t*)alloc((size_t)HID * 288 * 2);
    bf16_t* pkO   = (bf16_t*)alloc((size_t)HID * HID * 2);
    bf16_t* pk1   = (bf16_t*)alloc((size_t)HID * FFN_DIM * 2);
    bf16_t* pk2   = (bf16_t*)alloc((size_t)FFN_DIM * HID * 2);
    float*  b288  = (float*) alloc(288 * 4);
    (void)ws_size; (void)n_in; (void)out_size;

    const int TPB = 256;
    const int perW = HID * HID;            // 9216 packed halves per 96x96 weight
    const int regionQKV = (HID / 32) * 512 * (HID / 16);  // packed halves per 96x96

    // ---- weight repack (bf16, WMMA B-fragment layout) ----
    pack_w<<<(perW + TPB - 1) / TPB, TPB, 0, stream>>>(Wq, pkQKV, HID, HID);
    pack_w<<<(perW + TPB - 1) / TPB, TPB, 0, stream>>>(Wk, pkQKV + regionQKV, HID, HID);
    pack_w<<<(perW + TPB - 1) / TPB, TPB, 0, stream>>>(Wv, pkQKV + 2 * regionQKV, HID, HID);
    pack_w<<<(perW + TPB - 1) / TPB, TPB, 0, stream>>>(Wo, pkO, HID, HID);
    pack_w<<<(HID * FFN_DIM + TPB - 1) / TPB, TPB, 0, stream>>>(W1, pk1, HID, FFN_DIM);
    pack_w<<<(FFN_DIM * HID + TPB - 1) / TPB, TPB, 0, stream>>>(W2, pk2, FFN_DIM, HID);
    concat_bias3<<<1, 288, 0, stream>>>(bq, bk, bv, b288);

    // ---- attention block ----
    ln_kernel<<<(N + 7) / 8, TPB, 0, stream>>>(x, ln1_g, ln1_b, ybf, N);

    const int gemmBlocks = (N + 127) / 128;
    gemm_wmma<3, 0><<<gemmBlocks, TPB, 0, stream>>>(ybf, pkQKV, b288, nullptr,
                                                    qkv, nullptr, N, 288);

    int accN4 = N * (HID + NHEAD) / 4;
    zero_f32<<<2048, TPB, 0, stream>>>((float4*)wV, accN4);  // wV,Z contiguous

    edge_kernel<<<(E * NHEAD + TPB - 1) / TPB, TPB, 0, stream>>>(ei, qkv, wV, Zacc, E);
    attn_fin<<<(N * (HID / 4) + TPB - 1) / TPB, TPB, 0, stream>>>(wV, Zacc, abf, N);

    gemm_wmma<3, 1><<<gemmBlocks, TPB, 0, stream>>>(abf, pkO, bo, x,
                                                    x2, nullptr, N, HID);

    // ---- FFN block ----
    ln_kernel<<<(N + 7) / 8, TPB, 0, stream>>>(x2, ln2_g, ln2_b, y2bf, N);
    gemm_wmma<3, 2><<<gemmBlocks, TPB, 0, stream>>>(y2bf, pk1, b1, nullptr,
                                                    nullptr, g1bf, N, FFN_DIM);
    gemm_wmma<12, 1><<<gemmBlocks, TPB, 0, stream>>>(g1bf, pk2, b2, x2,
                                                     out, nullptr, N, HID);
}
